// MultiQueryAttention_28140625724106
// MI455X (gfx1250) — matrix operand order repaired and checked by reading, not compiled, not run on hardware
//
#include <hip/hip_runtime.h>
#include <hip/hip_bf16.h>
#include <math.h>

#define B_     2
#define NCTX   2048
#define DIM_   1024
#define NH     16
#define DH     64
#define INNER_ 1024

typedef __attribute__((ext_vector_type(16))) __bf16 v16bf;
typedef __attribute__((ext_vector_type(8)))  float  v8f;

// ---------------------------------------------------------------- helpers
static __device__ __forceinline__ unsigned short f2bf(float f) {
    unsigned u = __builtin_bit_cast(unsigned, f);
    unsigned r = (u + 0x7FFFu + ((u >> 16) & 1u)) >> 16;
    return (unsigned short)r;
}

static __device__ __forceinline__ v8f wmma_bf16(v16bf a, v16bf b, v8f c) {
    // D = A(16x32 bf16) x B(32x16 bf16) + C(16x16 f32)
    return __builtin_amdgcn_wmma_f32_16x16x32_bf16(
        /*neg_a=*/false, a, /*neg_b=*/false, b,
        /*c_mod=*/(short)0, c, /*reuse_a=*/false, /*reuse_b=*/false);
}

// A fragment (16x32 bf16) from row-major [.. , ld] source.
// lane = lo+16*hi : row = row0+lo ; elem e -> K = ((e>>3)<<4) + 8*hi + (e&7)
static __device__ __forceinline__ v16bf load_a(const unsigned short* base, int ld,
                                               int row0, int k0, int lane) {
    int lo = lane & 15, hi = lane >> 4;
    const __bf16* p = reinterpret_cast<const __bf16*>(
        base + (size_t)(row0 + lo) * ld + k0 + hi * 8);
    v16bf a;
#pragma unroll
    for (int e = 0; e < 8; ++e) a[e] = p[e];
#pragma unroll
    for (int e = 0; e < 8; ++e) a[8 + e] = p[16 + e];
    return a;
}

// B fragment (32x16 bf16): lane = lo+16*hi : col = col0+lo ; elem e -> K = ((e>>3)<<4) + 8*hi + (e&7)
static __device__ __forceinline__ v16bf load_b(const unsigned short* base, int ld,
                                               int k0, int col0, int lane) {
    int lo = lane & 15, hi = lane >> 4;
    const __bf16* p = reinterpret_cast<const __bf16*>(
        base + (size_t)(k0 + hi * 8) * ld + col0 + lo);
    v16bf b;
#pragma unroll
    for (int e = 0; e < 8; ++e) b[e] = p[(size_t)e * ld];
#pragma unroll
    for (int e = 0; e < 8; ++e) b[8 + e] = p[(size_t)(16 + e) * ld];
    return b;
}

// ---------------------------------------------------------------- kernel 1: pack f32 -> bf16
__global__ __launch_bounds__(256) void mqa_cvt_kernel(const float* __restrict__ src,
                                                      unsigned short* __restrict__ dst, int n) {
    int i = blockIdx.x * blockDim.x + threadIdx.x;
    int stride = gridDim.x * blockDim.x;
    for (; i < n; i += stride) dst[i] = f2bf(src[i]);
}

// ---------------------------------------------------------------- kernel 2: QKV proj + RoPE
// grid = (256 row-tiles of 16, 18 col-groups of 64), block = 1 wave.
// group g: 0..15 -> Q head g (rope + scale), 16 -> K (rope, store transposed), 17 -> V
__global__ __launch_bounds__(32) void mqa_qkv_rope_kernel(
        const unsigned short* __restrict__ xh,
        const unsigned short* __restrict__ Wqh,
        const unsigned short* __restrict__ Wkvh,
        unsigned short* __restrict__ Qh,    // [B, NH, NCTX, DH]
        unsigned short* __restrict__ KhT,   // [B, DH, NCTX]  (transposed)
        unsigned short* __restrict__ Vh) {  // [B, NCTX, DH]
    int lane = threadIdx.x & 31;
    int lo = lane & 15, hi = lane >> 4;
    int rt = blockIdx.x;
    int g  = blockIdx.y;
    int row0 = rt * 16;

    const unsigned short* W;
    int ldW, col0;
    if (g < 16) { W = Wqh;  ldW = INNER_; col0 = g * 64; }
    else        { W = Wkvh; ldW = 2*DH;   col0 = (g - 16) * 64; }

    v8f acc[4] = {};
    for (int k0 = 0; k0 < DIM_; k0 += 32) {
        if (k0 + 32 < DIM_) {
            __builtin_prefetch(xh + (size_t)(row0 + lo) * DIM_ + k0 + 32, 0, 0);
            __builtin_prefetch(W  + (size_t)(k0 + 32 + lane) * ldW + col0, 0, 0);
        }
        v16bf a = load_a(xh, DIM_, row0, k0, lane);
#pragma unroll
        for (int j = 0; j < 4; ++j) {
            v16bf b = load_b(W, ldW, k0, col0 + 16 * j, lane);
            acc[j] = wmma_bf16(a, b, acc[j]);
        }
    }

    // RoPE for Q and K. Partner of feature d is d+-32 = same lane/elem, acc[j] <-> acc[j+2].
    if (g <= 16) {
#pragma unroll
        for (int j = 0; j < 2; ++j) {
            float fi = (float)(16 * j + lo);                 // d % 32
            float invf = __expf(-0.2878231366f * fi);        // 10000^(-i/32)
#pragma unroll
            for (int e = 0; e < 8; ++e) {
                int gr = row0 + e + 8 * hi;
                float pos = (float)(gr & (NCTX - 1));
                float sn, cs;
                __sincosf(pos * invf, &sn, &cs);
                float x1 = acc[j][e], x2 = acc[j + 2][e];
                acc[j][e]     = x1 * cs - x2 * sn;
                acc[j + 2][e] = x2 * cs + x1 * sn;
            }
        }
    }
    float scale = (g < 16) ? 0.125f : 1.0f;   // DH^-0.5 applied to q only

#pragma unroll
    for (int j = 0; j < 4; ++j) {
#pragma unroll
        for (int e = 0; e < 8; ++e) {
            int gr  = row0 + e + 8 * hi;
            int b   = gr >> 11;
            int pos = gr & (NCTX - 1);
            int d   = 16 * j + lo;
            unsigned short v = f2bf(acc[j][e] * scale);
            if (g < 16)
                Qh[((size_t)(b * NH + g) * NCTX + pos) * DH + d] = v;
            else if (g == 16)
                KhT[((size_t)b * DH + d) * NCTX + pos] = v;
            else
                Vh[((size_t)(b * NCTX + pos)) * DH + d] = v;
        }
    }
}

// ---------------------------------------------------------------- kernel 3: causal flash attention
// grid = (32 q-blocks of 64, NH, B_), block = 4 waves; wave owns a 16x64 Q tile.
__global__ __launch_bounds__(128) void mqa_flash_attn_kernel(
        const unsigned short* __restrict__ Qh,   // [B, NH, NCTX, DH]
        const unsigned short* __restrict__ KhT,  // [B, DH, NCTX]
        const unsigned short* __restrict__ Vh,   // [B, NCTX, DH]
        unsigned short* __restrict__ Ah) {       // [B, NCTX, NH*DH]
    __shared__ unsigned short smem[4][16 * 32];  // per-wave P transpose staging

    int lane = threadIdx.x & 31;
    int w    = threadIdx.x >> 5;
    int lo = lane & 15, hi = lane >> 4;
    int bx = blockIdx.x, h = blockIdx.y, b = blockIdx.z;

    const unsigned short* Qbh = Qh  + (size_t)(b * NH + h) * NCTX * DH;
    const unsigned short* Kb  = KhT + (size_t)b * DH * NCTX;
    const unsigned short* Vb  = Vh  + (size_t)b * NCTX * DH;

    int row0 = bx * 64 + w * 16;   // q rows of this wave
    v16bf qa0 = load_a(Qbh, DH, row0, 0,  lane);
    v16bf qa1 = load_a(Qbh, DH, row0, 32, lane);

    v8f o[4] = {};
    float m[8], l[8];
#pragma unroll
    for (int e = 0; e < 8; ++e) { m[e] = -3.0e38f; l[e] = 0.0f; }

    int jend = bx * 64 + 64;                     // causal bound (exclusive)
    for (int j0 = 0; j0 < jend; j0 += 32) {
        // ---- S = q . k^T  (two 16-col tiles)
        v8f s[2];
#pragma unroll
        for (int jt = 0; jt < 2; ++jt) {
            v8f a = {};
            v16bf kb0 = load_b(Kb, NCTX, 0,  j0 + 16 * jt, lane);
            a = wmma_bf16(qa0, kb0, a);
            v16bf kb1 = load_b(Kb, NCTX, 32, j0 + 16 * jt, lane);
            a = wmma_bf16(qa1, kb1, a);
            s[jt] = a;
        }
        // ---- causal mask + running row max (row lives on 16 lanes of one half)
        float newm[8];
#pragma unroll
        for (int e = 0; e < 8; ++e) {
            int gi = row0 + e + 8 * hi;
#pragma unroll
            for (int jt = 0; jt < 2; ++jt) {
                int gj = j0 + 16 * jt + lo;
                if (gj > gi) s[jt][e] = -3.0e38f;
            }
            newm[e] = fmaxf(s[0][e], s[1][e]);
        }
#pragma unroll
        for (int mask = 1; mask < 16; mask <<= 1)
#pragma unroll
            for (int e = 0; e < 8; ++e)
                newm[e] = fmaxf(newm[e], __shfl_xor(newm[e], mask, 32));
#pragma unroll
        for (int e = 0; e < 8; ++e) newm[e] = fmaxf(newm[e], m[e]);

        // ---- P = exp(s - newm), row sums, rescale running state
        float rs[8];
#pragma unroll
        for (int e = 0; e < 8; ++e) {
            float p0 = __expf(s[0][e] - newm[e]);
            float p1 = __expf(s[1][e] - newm[e]);
            s[0][e] = p0; s[1][e] = p1;
            rs[e] = p0 + p1;
        }
#pragma unroll
        for (int mask = 1; mask < 16; mask <<= 1)
#pragma unroll
            for (int e = 0; e < 8; ++e)
                rs[e] += __shfl_xor(rs[e], mask, 32);
#pragma unroll
        for (int e = 0; e < 8; ++e) {
            float alpha = __expf(m[e] - newm[e]);
            l[e] = l[e] * alpha + rs[e];
            m[e] = newm[e];
#pragma unroll
            for (int t = 0; t < 4; ++t) o[t][e] *= alpha;
        }

        // ---- P: D-layout f32 -> A-layout bf16 via per-wave LDS tile (16x32, ld=32)
        unsigned short* pl = smem[w];
#pragma unroll
        for (int jt = 0; jt < 2; ++jt)
#pragma unroll
            for (int e = 0; e < 8; ++e)
                pl[(e + 8 * hi) * 32 + 16 * jt + lo] = f2bf(s[jt][e]);
        asm volatile("s_wait_dscnt 0" ::: "memory");
        v16bf pa = load_a(pl, 32, 0, 0, lane);

        // ---- O += P . V
#pragma unroll
        for (int t = 0; t < 4; ++t) {
            v16bf vb = load_b(Vb, DH, j0, 16 * t, lane);
            o[t] = wmma_bf16(pa, vb, o[t]);
        }
    }

    // ---- epilogue: normalize, store [b, pos, h*64 + d] as bf16
#pragma unroll
    for (int e = 0; e < 8; ++e) {
        int grow = row0 + e + 8 * hi;
        float inv = 1.0f / l[e];
        size_t rowbase = ((size_t)(b * NCTX + grow) * NH + h) * DH;
#pragma unroll
        for (int t = 0; t < 4; ++t)
            Ah[rowbase + 16 * t + lo] = f2bf(o[t][e] * inv);
    }
}

// ---------------------------------------------------------------- kernel 4: output projection (f32 out)
// grid = (256 row-tiles, 16 col-groups of 64), block = 1 wave.
__global__ __launch_bounds__(32) void mqa_oproj_kernel(
        const unsigned short* __restrict__ Ah,
        const unsigned short* __restrict__ Woh,
        float* __restrict__ out) {
    int lane = threadIdx.x & 31;
    int lo = lane & 15, hi = lane >> 4;
    int rt = blockIdx.x;
    int col0 = blockIdx.y * 64;
    int row0 = rt * 16;

    v8f acc[4] = {};
    for (int k0 = 0; k0 < INNER_; k0 += 32) {
        if (k0 + 32 < INNER_) {
            __builtin_prefetch(Ah  + (size_t)(row0 + lo) * INNER_ + k0 + 32, 0, 0);
            __builtin_prefetch(Woh + (size_t)(k0 + 32 + lane) * DIM_ + col0, 0, 0);
        }
        v16bf a = load_a(Ah, INNER_, row0, k0, lane);
#pragma unroll
        for (int j = 0; j < 4; ++j) {
            v16bf b = load_b(Woh, DIM_, k0, col0 + 16 * j, lane);
            acc[j] = wmma_bf16(a, b, acc[j]);
        }
    }
#pragma unroll
    for (int j = 0; j < 4; ++j)
#pragma unroll
        for (int e = 0; e < 8; ++e)
            out[(size_t)(row0 + e + 8 * hi) * DIM_ + col0 + 16 * j + lo] = acc[j][e];
}

// ---------------------------------------------------------------- launch
extern "C" void kernel_launch(void* const* d_in, const int* in_sizes, int n_in,
                              void* d_out, int out_size, void* d_ws, size_t ws_size,
                              hipStream_t stream) {
    (void)in_sizes; (void)n_in; (void)out_size; (void)ws_size;
    const float* x   = (const float*)d_in[0];
    const float* Wq  = (const float*)d_in[1];
    const float* Wkv = (const float*)d_in[2];
    const float* Wo  = (const float*)d_in[3];
    float* out = (float*)d_out;

    char* ws = (char*)d_ws;
    size_t off = 0;
    auto alloc = [&](size_t elems) -> unsigned short* {
        unsigned short* p = (unsigned short*)(ws + off);
        off += ((elems * 2 + 255) / 256) * 256;
        return p;
    };
    const size_t nX   = (size_t)B_ * NCTX * DIM_;     // 4,194,304
    const size_t nWq  = (size_t)DIM_ * INNER_;        // 1,048,576
    const size_t nWkv = (size_t)DIM_ * 2 * DH;        //   131,072
    const size_t nWo  = (size_t)INNER_ * DIM_;        // 1,048,576
    unsigned short* xh   = alloc(nX);
    unsigned short* Wqh  = alloc(nWq);
    unsigned short* Wkvh = alloc(nWkv);
    unsigned short* Woh  = alloc(nWo);
    unsigned short* Qh   = alloc((size_t)B_ * NH * NCTX * DH);
    unsigned short* KhT  = alloc((size_t)B_ * DH * NCTX);
    unsigned short* Vh   = alloc((size_t)B_ * NCTX * DH);
    unsigned short* Ah   = alloc((size_t)B_ * NCTX * INNER_);

    mqa_cvt_kernel<<<1024, 256, 0, stream>>>(x,   xh,   (int)nX);
    mqa_cvt_kernel<<<512,  256, 0, stream>>>(Wq,  Wqh,  (int)nWq);
    mqa_cvt_kernel<<<64,   256, 0, stream>>>(Wkv, Wkvh, (int)nWkv);
    mqa_cvt_kernel<<<512,  256, 0, stream>>>(Wo,  Woh,  (int)nWo);

    mqa_qkv_rope_kernel<<<dim3(256, 18), 32, 0, stream>>>(xh, Wqh, Wkvh, Qh, KhT, Vh);
    mqa_flash_attn_kernel<<<dim3(32, NH, B_), 128, 0, stream>>>(Qh, KhT, Vh, Ah);
    mqa_oproj_kernel<<<dim3(256, 16), 32, 0, stream>>>(Ah, Woh, out);
}
